// LSTMDecoder_17763984736678
// MI455X (gfx1250) — matrix operand order repaired and checked by reading, not compiled, not run on hardware
//
#include <hip/hip_runtime.h>
#include <hip/hip_bf16.h>
#include <cmath>

// ---------------- problem dimensions ----------------
constexpr int BB  = 16;     // batch
constexpr int TT  = 64;     // time steps
constexpr int SS  = 100;    // source length
constexpr int VV  = 32000;  // vocab
constexpr int EE  = 300;    // embed dim
constexpr int HH  = 1024;   // hidden
constexpr int G4H = 4 * HH; // 4096 gate width
// K-concat layouts (all K multiples of 32 so the WMMA K-loop is exact):
constexpr int KPAD0 = 1344;        // (E + H)=1324 padded to 32-multiple
constexpr int K0    = KPAD0 + HH;  // 2368 : A = [x | feed | pad | h1]
constexpr int K1    = 2 * HH;      // 2048 : A = [h1 | h2] (and [ctx | h2])

typedef __attribute__((ext_vector_type(16))) __bf16 v16bf;
typedef __attribute__((ext_vector_type(8)))  __bf16 v8bf;
typedef __attribute__((ext_vector_type(8)))  float  v8f;

// ---------------- CDNA5 async / wait helpers ----------------
__device__ __forceinline__ void wait_async_le2() {
#if __has_builtin(__builtin_amdgcn_s_wait_asynccnt)
  __builtin_amdgcn_s_wait_asynccnt(2);
#else
  asm volatile("s_wait_asynccnt 0x2" ::: "memory");
#endif
}
__device__ __forceinline__ void wait_async_le0() {
#if __has_builtin(__builtin_amdgcn_s_wait_asynccnt)
  __builtin_amdgcn_s_wait_asynccnt(0);
#else
  asm volatile("s_wait_asynccnt 0x0" ::: "memory");
#endif
}
// One b128 global->LDS async transfer per lane (VGLOBAL async encoding:
// vdst = LDS byte address, vaddr = 64-bit global address). ASYNCcnt-tracked.
__device__ __forceinline__ void async_cp16(unsigned lds_addr, const void* gsrc) {
  asm volatile("global_load_async_to_lds_b128 %0, %1, off"
               :: "v"(lds_addr), "v"(gsrc) : "memory");
}

// ---------------- WMMA fragment loaders (contiguous k order: k = 16*(lane/16) + s) ----
// A 16x32 (MxK) from LDS tile (row-major 16x32 bf16, 64B rows):
// lanes 0-15 row M=lane hold K 0..15; lanes 16-31 hold K 16..31.
__device__ __forceinline__ v16bf fragA_lds(const __bf16* __restrict__ tile, int lane) {
  const int row  = lane & 15;
  const int kblk = (lane >> 4) << 4; // 0 or 16
  const __bf16* p = tile + row * 32 + kblk;
  v8bf lo = *(const v8bf*)(p);       // K = kblk .. kblk+7   (ds_load_b128)
  v8bf hi = *(const v8bf*)(p + 8);   // K = kblk+8 .. kblk+15
  v16bf f;
#pragma unroll
  for (int i = 0; i < 8; ++i) { f[i] = lo[i]; f[i + 8] = hi[i]; }
  return f;
}
// B 32x16 (KxN) loaded from row-major W[N,K] (we compute X @ W^T):
// lanes 0-15 col N=lane hold K 0..15; lanes 16-31 col N=lane-16 hold K 16..31.
__device__ __forceinline__ v16bf load_fragB(const __bf16* __restrict__ base,
                                            int ld, int lane) {
  const int col  = lane & 15;
  const int kblk = (lane >> 4) << 4; // 0 or 16
  const __bf16* p = base + (long long)col * ld + kblk;
  v8bf lo = *(const v8bf*)(p);
  v8bf hi = *(const v8bf*)(p + 8);
  v16bf f;
#pragma unroll
  for (int i = 0; i < 8; ++i) { f[i] = lo[i]; f[i + 8] = hi[i]; }
  return f;
}

// ---------------- generic bf16 WMMA GEMM: C[M,N] = A[M,K] @ W[N,K]^T --------
// 256 threads = 8 waves; each wave owns a 16x32 output tile. The block-shared
// 16x32 A tile is double-buffered in LDS via global_load_async_to_lds_b128
// (wave 0 issues, s_wait_asynccnt + barrier publishes). B fragments are
// software-pipelined one K-step ahead so s_wait_loadcnt overlaps the WMMAs.
// N must be a multiple of 256, M of 16, K of 32.
__global__ __launch_bounds__(256)
void k_gemm(const __bf16* __restrict__ A, int lda,
            const __bf16* __restrict__ W, int ldw,
            const float* __restrict__ bias,
            float* __restrict__ Cf, long long ldc,
            __bf16* __restrict__ Cb, long long ldcb,
            __bf16* __restrict__ Cb2, long long ldcb2,
            int N, int K, int act) {
  __shared__ __bf16 sA[2][16 * 32] __attribute__((aligned(16)));
  const int lane = threadIdx.x & 31;
  const int wave = threadIdx.x >> 5;
  const int m0 = blockIdx.y << 4;
  const int n0 = (blockIdx.x * 8 + wave) << 5;   // 32 cols per wave
  if (n0 >= N) return;                            // wave-uniform
  const __bf16* Ap  = A + (long long)m0 * lda;
  const __bf16* Wp0 = W + (long long)n0 * ldw;
  const __bf16* Wp1 = W + (long long)(n0 + 16) * ldw;

  // wave 0: stage the 16x32 bf16 A tile (16 rows x 64B) with 2 async b128/lane
  auto issueA = [&](int bufi, int k) {
#pragma unroll
    for (int half = 0; half < 2; ++half) {
      const int j = lane + half * 32;           // 0..63 transfers of 16B
      const int row = j >> 2, q = j & 3;
      const __bf16* src = Ap + (long long)row * lda + k + q * 8;
      const unsigned dst = (unsigned)(uintptr_t)(&sA[bufi][row * 32 + q * 8]);
      async_cp16(dst, (const void*)src);
    }
  };

  v8f acc0 = {}; v8f acc1 = {};
  if (wave == 0) issueA(0, 0);
  v16bf b0 = load_fragB(Wp0, ldw, lane);          // B pipeline prologue
  v16bf b1 = load_fragB(Wp1, ldw, lane);
  int buf = 0;
  for (int k = 0; k < K; k += 32) {
    const bool more = (k + 32) < K;
    if (wave == 0) {
      if (more) { issueA(buf ^ 1, k + 32); wait_async_le2(); }
      else      { wait_async_le0(); }
    }
    __syncthreads();                              // publish sA[buf]
    v16bf a = fragA_lds(&sA[buf][0], lane);
    v16bf nb0 = b0, nb1 = b1;
    if (more) {
      if (k + 160 < K) {                          // weight stream prefetch
        __builtin_prefetch(Wp0 + k + 160, 0, 1);
        __builtin_prefetch(Wp1 + k + 160, 0, 1);
      }
      nb0 = load_fragB(Wp0 + k + 32, ldw, lane);  // loads overlap WMMAs below
      nb1 = load_fragB(Wp1 + k + 32, ldw, lane);
    }
    acc0 = __builtin_amdgcn_wmma_f32_16x16x32_bf16(false, a, false, b0,
                                                   (short)0, acc0, false, false);
    acc1 = __builtin_amdgcn_wmma_f32_16x16x32_bf16(false, a, false, b1,
                                                   (short)0, acc1, false, false);
    b0 = nb0; b1 = nb1;
    __syncthreads();                              // sA[buf] free for rewrite
    buf ^= 1;
  }
  // C/D layout: VGPR i -> M = i + 8*(lane>=16), N = lane&15  (ISA 7.12.2)
  const int mrow = (lane >> 4) << 3;
  const int ncol = lane & 15;
#pragma unroll
  for (int i = 0; i < 8; ++i) {
    const int m  = m0 + mrow + i;
    const int na = n0 + ncol;
    const int nb = n0 + 16 + ncol;
    float v0 = acc0[i] + (bias ? bias[na] : 0.0f);
    float v1 = acc1[i] + (bias ? bias[nb] : 0.0f);
    if (act == 1) { v0 = tanhf(v0); v1 = tanhf(v1); }
    if (Cf)  { Cf [(long long)m * ldc  + na] = v0;         Cf [(long long)m * ldc  + nb] = v1; }
    if (Cb)  { Cb [(long long)m * ldcb + na] = (__bf16)v0; Cb [(long long)m * ldcb + nb] = (__bf16)v1; }
    if (Cb2) { Cb2[(long long)m * ldcb2+ na] = (__bf16)v0; Cb2[(long long)m * ldcb2+ nb] = (__bf16)v1; }
  }
}

// ---------------- pointwise / setup kernels ----------------
__device__ __forceinline__ float sigmoidf(float x) { return 1.0f / (1.0f + expf(-x)); }

// LSTM cell update from fused gates G[16,4096] = [i|f|g|o]; c updated in place
// (f32); new h written as bf16 to up to two activation slots.
__global__ __launch_bounds__(256)
void k_lstm(const float* __restrict__ G, float* __restrict__ c,
            __bf16* __restrict__ d1, long long ld1,
            __bf16* __restrict__ d2, long long ld2) {
  const int idx = blockIdx.x * 256 + threadIdx.x;  // BB*HH
  const int b = idx >> 10, h = idx & (HH - 1);
  const float* g = G + (long long)b * G4H;
  const float i = sigmoidf(g[h]);
  const float f = sigmoidf(g[HH + h]);
  const float gg = tanhf(g[2 * HH + h]);
  const float o = sigmoidf(g[3 * HH + h]);
  const float cn = f * c[idx] + i * gg;
  c[idx] = cn;
  const float hn = o * tanhf(cn);
  d1[(long long)b * ld1 + h] = (__bf16)hn;
  if (d2) d2[(long long)b * ld2 + h] = (__bf16)hn;
}

// Gather x_t = embed[prev_tokens[:,t]] into actA cols [0,E)
__global__ __launch_bounds__(256)
void k_stage_x(const int* __restrict__ tw, int t,
               const float* __restrict__ embed, __bf16* __restrict__ actA) {
  const int idx = blockIdx.x * 256 + threadIdx.x;
  if (idx >= BB * EE) return;
  const int b = idx / EE, e = idx - b * EE;
  const int tok = tw[b * (TT + 1) + t];
  actA[(long long)b * K0 + e] = (__bf16)embed[(long long)tok * EE + e];
}

// Initial cell states: c1 = c2 = masked max over S of encoder_outs
__global__ __launch_bounds__(256)
void k_ctx0(const float* __restrict__ enc, const int* __restrict__ mask,
            float* __restrict__ c1, float* __restrict__ c2) {
  const int idx = blockIdx.x * 256 + threadIdx.x;  // BB*HH
  const int b = idx >> 10, h = idx & (HH - 1);
  float m = -__builtin_inff();
  for (int s = 0; s < SS; ++s)
    if (mask[b * SS + s])
      m = fmaxf(m, enc[((long long)(b * SS + s)) * HH + h]);
  c1[idx] = m; c2[idx] = m;
}

// Attention for one batch row per block: scores, masked softmax, context.
__global__ __launch_bounds__(256)
void k_attn(const float* __restrict__ q, const float* __restrict__ enc,
            const int* __restrict__ mask, __bf16* __restrict__ ctx, long long ldctx) {
  __shared__ float sq[HH];
  __shared__ float sa[128];
  __shared__ float ssum;
  const int b = blockIdx.x, tid = threadIdx.x;
  const int wave = tid >> 5, lane = tid & 31;
  for (int h = tid; h < HH; h += 256) sq[h] = q[b * HH + h];
  for (int s = tid; s < 128; s += 256) sa[s] = -__builtin_inff();
  __syncthreads();
  for (int s = wave; s < SS; s += 8) {           // one wave per source pos
    const float* e = enc + ((long long)(b * SS + s)) * HH;
    float p = 0.0f;
    for (int h = lane; h < HH; h += 32) p += e[h] * sq[h];
    for (int off = 16; off > 0; off >>= 1) p += __shfl_down(p, off, 32);
    if (lane == 0) sa[s] = (mask[b * SS + s] != 0) ? p : -__builtin_inff();
  }
  __syncthreads();
  if (wave == 0) {                                // softmax over S=100
    float mx = -__builtin_inff();
    for (int s = lane; s < SS; s += 32) mx = fmaxf(mx, sa[s]);
    for (int off = 16; off > 0; off >>= 1) mx = fmaxf(mx, __shfl_down(mx, off, 32));
    mx = __shfl(mx, 0, 32);
    float sum = 0.0f;
    for (int s = lane; s < SS; s += 32) { float e = expf(sa[s] - mx); sa[s] = e; sum += e; }
    for (int off = 16; off > 0; off >>= 1) sum += __shfl_down(sum, off, 32);
    if (lane == 0) ssum = sum;
  }
  __syncthreads();
  const float inv = 1.0f / ssum;
  for (int h = tid; h < HH; h += 256) {
    float a = 0.0f;
    for (int s = 0; s < SS; ++s) a += sa[s] * enc[((long long)(b * SS + s)) * HH + h];
    ctx[(long long)b * ldctx + h] = (__bf16)(a * inv);
  }
}

// Per-row log-softmax NLL accumulation (row r = b*T + t)
__global__ __launch_bounds__(256)
void k_nll(const float* __restrict__ logits, const int* __restrict__ tw,
           float* __restrict__ acc) {
  __shared__ float sm[256];
  const int r = blockIdx.x, tid = threadIdx.x;
  const float* row = logits + (long long)r * VV;
  float mx = -__builtin_inff();
  for (int v = tid; v < VV; v += 256) mx = fmaxf(mx, row[v]);
  sm[tid] = mx; __syncthreads();
  for (int s = 128; s > 0; s >>= 1) { if (tid < s) sm[tid] = fmaxf(sm[tid], sm[tid + s]); __syncthreads(); }
  mx = sm[0]; __syncthreads();
  float sum = 0.0f;
  for (int v = tid; v < VV; v += 256) sum += expf(row[v] - mx);
  sm[tid] = sum; __syncthreads();
  for (int s = 128; s > 0; s >>= 1) { if (tid < s) sm[tid] += sm[tid + s]; __syncthreads(); }
  if (tid == 0) {
    const int b = r >> 6, t = r & 63;
    const int tgt = tw[b * (TT + 1) + t + 1];
    const float nll = (mx + logf(sm[0])) - row[tgt];
    if (tgt != 0) { atomicAdd(&acc[0], nll); atomicAdd(&acc[1], 1.0f); }
  }
}

__global__ void k_loss(const float* __restrict__ acc, float* __restrict__ out) {
  out[0] = acc[0] / fmaxf(acc[1], 1.0f);
}

// f32 -> bf16 weight conversion with K-concat + zero padding:
// dst is N x (kaPad+kb); cols [0,ka) from A, [ka,kaPad) zero, [kaPad,..) from Bs.
__global__ __launch_bounds__(256)
void k_cvt(const float* __restrict__ A, int ka, int kaPad,
           const float* __restrict__ Bs, int kb,
           __bf16* __restrict__ dst, long long total) {
  const int ld = kaPad + kb;
  const long long stride = (long long)gridDim.x * 256;
  for (long long i = (long long)blockIdx.x * 256 + threadIdx.x; i < total; i += stride) {
    const long long n = i / ld; const int k = (int)(i - n * ld);
    float v;
    if (k < kaPad) v = (k < ka) ? A[n * ka + k] : 0.0f;
    else           v = Bs[n * kb + (k - kaPad)];
    dst[i] = (__bf16)v;
  }
}

__global__ __launch_bounds__(256)
void k_bsum(const float* __restrict__ a, const float* __restrict__ b,
            float* __restrict__ o, int n) {
  const int i = blockIdx.x * 256 + threadIdx.x;
  if (i < n) o[i] = a[i] + b[i];
}

__global__ __launch_bounds__(256)
void k_zero(unsigned int* __restrict__ p, long long n32) {
  const long long stride = (long long)gridDim.x * 256;
  for (long long i = (long long)blockIdx.x * 256 + threadIdx.x; i < n32; i += stride)
    p[i] = 0u;
}

// ---------------- host driver ----------------
extern "C" void kernel_launch(void* const* d_in, const int* in_sizes, int n_in,
                              void* d_out, int out_size, void* d_ws, size_t ws_size,
                              hipStream_t stream) {
  (void)in_sizes; (void)n_in; (void)out_size; (void)ws_size;
  const int*   tw    = (const int*)  d_in[0];
  const float* enc   = (const float*)d_in[1];
  const int*   mask  = (const int*)  d_in[2];
  const float* embed = (const float*)d_in[3];
  const float* W_ih0 = (const float*)d_in[4];
  const float* W_hh0 = (const float*)d_in[5];
  const float* b_ih0 = (const float*)d_in[6];
  const float* b_hh0 = (const float*)d_in[7];
  const float* W_ih1 = (const float*)d_in[8];
  const float* W_hh1 = (const float*)d_in[9];
  const float* b_ih1 = (const float*)d_in[10];
  const float* b_hh1 = (const float*)d_in[11];
  const float* W_ai  = (const float*)d_in[12];
  const float* W_ao  = (const float*)d_in[13];
  const float* W_out = (const float*)d_in[14];
  const float* b_out = (const float*)d_in[15];
  float* logits = (float*)d_out;

  // workspace carve-out (~111 MB, 256B aligned slabs)
  char* cur = (char*)d_ws;
  auto take = [&](size_t bytes) -> void* {
    void* p = (void*)cur; cur += (bytes + 255) & ~(size_t)255; return p;
  };
  __bf16* Wc0  = (__bf16*)take((size_t)G4H * K0 * 2);       // [W_ih0|0|W_hh0]
  __bf16* Wc1  = (__bf16*)take((size_t)G4H * K1 * 2);       // [W_ih1|W_hh1]
  __bf16* Wai  = (__bf16*)take((size_t)HH * HH * 2);
  __bf16* Wao  = (__bf16*)take((size_t)HH * 2 * HH * 2);
  __bf16* Wo   = (__bf16*)take((size_t)VV * HH * 2);
  float*  bs0  = (float*) take((size_t)G4H * 4);
  float*  bs1  = (float*) take((size_t)G4H * 4);
  __bf16* actA = (__bf16*)take((size_t)BB * K0 * 2);        // [x|feed|pad|h1]
  __bf16* actB = (__bf16*)take((size_t)BB * K1 * 2);        // [h1|h2]
  __bf16* actC = (__bf16*)take((size_t)BB * K1 * 2);        // [ctx|h2]
  float*  Gbuf = (float*) take((size_t)BB * G4H * 4);
  float*  qbuf = (float*) take((size_t)BB * HH * 4);
  float*  c1   = (float*) take((size_t)BB * HH * 4);
  float*  c2   = (float*) take((size_t)BB * HH * 4);
  __bf16* Abig = (__bf16*)take((size_t)BB * TT * HH * 2);   // rows r=b*T+t
  float*  acc  = (float*) take(256);

  const dim3 blk(256);
  // one-time (per call) weight prep: bf16 convert + K-concat + bias fold
  k_cvt<<<dim3(2048), blk, 0, stream>>>(W_ih0, EE + HH, KPAD0, W_hh0, HH, Wc0, (long long)G4H * K0);
  k_cvt<<<dim3(2048), blk, 0, stream>>>(W_ih1, HH, HH, W_hh1, HH, Wc1, (long long)G4H * K1);
  k_cvt<<<dim3(1024), blk, 0, stream>>>(W_ai, HH, HH, nullptr, 0, Wai, (long long)HH * HH);
  k_cvt<<<dim3(1024), blk, 0, stream>>>(W_ao, 2 * HH, 2 * HH, nullptr, 0, Wao, (long long)HH * 2 * HH);
  k_cvt<<<dim3(4096), blk, 0, stream>>>(W_out, HH, HH, nullptr, 0, Wo, (long long)VV * HH);
  k_bsum<<<dim3(16), blk, 0, stream>>>(b_ih0, b_hh0, bs0, G4H);
  k_bsum<<<dim3(16), blk, 0, stream>>>(b_ih1, b_hh1, bs1, G4H);
  // state init (deterministic every call)
  k_zero<<<dim3(64), blk, 0, stream>>>((unsigned*)actA, (long long)BB * K0 / 2);
  k_zero<<<dim3(64), blk, 0, stream>>>((unsigned*)actB, (long long)BB * K1 / 2);
  k_zero<<<dim3(64), blk, 0, stream>>>((unsigned*)actC, (long long)BB * K1 / 2);
  k_zero<<<dim3(1),  blk, 0, stream>>>((unsigned*)acc, 64);
  k_ctx0<<<dim3(BB * HH / 256), blk, 0, stream>>>(enc, mask, c1, c2);

  // recurrent scan: weights stay resident in the 192MB L2 across all 64 steps
  for (int t = 0; t < TT; ++t) {
    k_stage_x<<<dim3((BB * EE + 255) / 256), blk, 0, stream>>>(tw, t, embed, actA);
    // gates0 = [x|feed|h1] @ [W_ih0|W_hh0]^T + (b_ih0+b_hh0)
    k_gemm<<<dim3(G4H / 256, 1), blk, 0, stream>>>(actA, K0, Wc0, K0, bs0,
        Gbuf, G4H, nullptr, 0, nullptr, 0, G4H, K0, 0);
    k_lstm<<<dim3(BB * HH / 256), blk, 0, stream>>>(Gbuf, c1, actA + KPAD0, K0, actB, K1);
    // gates1 = [h1|h2_prev] @ [W_ih1|W_hh1]^T + (b_ih1+b_hh1)
    k_gemm<<<dim3(G4H / 256, 1), blk, 0, stream>>>(actB, K1, Wc1, K1, bs1,
        Gbuf, G4H, nullptr, 0, nullptr, 0, G4H, K1, 0);
    k_lstm<<<dim3(BB * HH / 256), blk, 0, stream>>>(Gbuf, c2, actB + HH, K1, actC + HH, K1);
    // q = h2 @ W_attn_in^T
    k_gemm<<<dim3(HH / 256, 1), blk, 0, stream>>>(actC + HH, K1, Wai, HH, nullptr,
        qbuf, HH, nullptr, 0, nullptr, 0, HH, HH, 0);
    k_attn<<<dim3(BB), blk, 0, stream>>>(qbuf, enc, mask, actC, K1);
    // out = tanh([ctx|h2] @ W_attn_out^T) -> feed slot of actA + row (b*T+t) of Abig
    k_gemm<<<dim3(HH / 256, 1), blk, 0, stream>>>(actC, K1, Wao, 2 * HH, nullptr,
        nullptr, 0, actA + EE, K0, Abig + (size_t)t * HH, (long long)TT * HH, HH, K1, 1);
  }

  // logits = outs @ W_out^T + b_out  (1024 x 32000, K=1024, bf16 WMMA, f32 acc)
  k_gemm<<<dim3(VV / 256, (BB * TT) / 16), blk, 0, stream>>>(Abig, HH, Wo, HH, b_out,
      logits, VV, nullptr, 0, nullptr, 0, VV, HH, 0);
  // masked mean NLL over log_softmax(logits)
  k_nll<<<dim3(BB * TT), blk, 0, stream>>>(logits, tw, acc);
  k_loss<<<dim3(1), dim3(1), 0, stream>>>(acc, logits + (size_t)BB * TT * VV);
}
